// GVP_86242943303736
// MI455X (gfx1250) — compile-verified, matrix operand order checked by reading, not run on hardware
//
#include <hip/hip_runtime.h>
#include <math.h>

typedef __bf16 bf16;
typedef __attribute__((ext_vector_type(16))) __bf16 v16bf;
typedef __attribute__((ext_vector_type(8)))  __bf16 v8bf;
typedef __attribute__((ext_vector_type(8)))  float   v8f;
typedef __attribute__((ext_vector_type(4)))  float   v4f;

#define WAVES_PER_WG 2
// per-wave LDS layout (bf16 element offsets); padded strides 72 / 136 keep the
// 16B DS chunk loads bank-conflict free (144B, 272B row strides).
#define SV_OFF   0        // 3 planes * 16 rows * 72  (V tile, bf16)
#define SV_PLANE 1152
#define SV_LD    72
#define VH_OFF   3456     // 3 planes * 16 rows * 136 (Vh, bf16)
#define VH_PLANE 2176
#define VH_LD    136
#define NRM_OFF  9984     // 16 * 136 (||Vh||)
#define SO_OFF   12160    // 16 * 136 (relu'd s_out, bf16)
#define WAVE_LDS 14336    // bf16 elements per wave (28672 B)

// ---- WMMA fragment loaders -------------------------------------------------
// A (16x32 bf16): lane holds row m=lane%16; K-chunks [kh*8, kh*8+8) and
// [16+kh*8, ...). B (32x16) is symmetric: lane holds out-col n=lane%16 == a row
// of the [out][in] weight matrix with the same K pattern.
static __device__ __forceinline__ v16bf frag_bf(const bf16* rowk, int kh) {
  v8bf c0 = *(const v8bf*)(rowk + kh * 8);
  v8bf c1 = *(const v8bf*)(rowk + 16 + kh * 8);
  v16bf a;
#pragma unroll
  for (int e = 0; e < 8; ++e) { a[e] = c0[e]; a[8 + e] = c1[e]; }
  return a;
}

static __device__ __forceinline__ v16bf frag_f32(const float* rowk, int kh) {
  v4f c0 = *(const v4f*)(rowk + kh * 8);
  v4f c1 = *(const v4f*)(rowk + kh * 8 + 4);
  v4f c2 = *(const v4f*)(rowk + 16 + kh * 8);
  v4f c3 = *(const v4f*)(rowk + 16 + kh * 8 + 4);
  v16bf a;
#pragma unroll
  for (int e = 0; e < 4; ++e) {
    a[e]      = (bf16)c0[e];
    a[4 + e]  = (bf16)c1[e];
    a[8 + e]  = (bf16)c2[e];
    a[12 + e] = (bf16)c3[e];
  }
  return a;
}

#define WMMA_BF16(A, B, C) \
  __builtin_amdgcn_wmma_f32_16x16x32_bf16(false, (A), false, (B), (short)0, (C), false, false)

// ---- weight f32 -> bf16 preconversion into d_ws (112 KB, L2-resident) ------
// ws layout (bf16 elems): WH 128x64 @0, WV 64x128 @8192, WS 128x256 @16384, WG 64x128 @49152
__global__ __launch_bounds__(256) void gvp_wconv(
    const float* __restrict__ wh, const float* __restrict__ wv,
    const float* __restrict__ ws, const float* __restrict__ wg,
    bf16* __restrict__ out) {
  int t = blockIdx.x * 256 + threadIdx.x;
  if (t < 8192)        out[t] = (bf16)wh[t];
  else if (t < 16384)  out[t] = (bf16)wv[t - 8192];
  else if (t < 49152)  out[t] = (bf16)ws[t - 16384];
  else if (t < 57344)  out[t] = (bf16)wg[t - 49152];
}

// ---- fused GVP kernel: one wave owns a 16-node M-tile ----------------------
__global__ __launch_bounds__(32 * WAVES_PER_WG) void gvp_fused(
    const float* __restrict__ sIn, const float* __restrict__ Vin,
    const bf16* __restrict__ WH, const bf16* __restrict__ WVm,
    const bf16* __restrict__ WS, const bf16* __restrict__ WG,
    const float* __restrict__ bS, const float* __restrict__ bG,
    float* __restrict__ sOut, float* __restrict__ vOut) {
  extern __shared__ bf16 ldsAll[];
  const int lane = threadIdx.x & 31;
  const int wave = threadIdx.x >> 5;
  bf16* lds = ldsAll + wave * WAVE_LDS;

  const int m  = lane & 15;   // A-fragment row
  const int kh = lane >> 4;   // A/B K-half select
  const int col = lane & 15;  // B out-col == D col
  const int hi  = lane >> 4;  // D row-half: rows m = r + 8*hi
  const long n0 = ((long)blockIdx.x * WAVES_PER_WG + wave) * 16;

  // ---- stage V tile [16 nodes][64 vi][3 c] f32 -> 3 bf16 LDS planes [16][64]
  {
    const float* Vt = Vin + n0 * 192;  // 3072 contiguous floats
#pragma unroll
    for (int i = 0; i < 24; ++i) {
      int q = i * 32 + lane;           // float4 index 0..767 (coalesced b128)
      v4f d = *(const v4f*)(Vt + q * 4);
#pragma unroll
      for (int j = 0; j < 4; ++j) {
        int e = q * 4 + j;
        int node = e / 192;
        int w = e - node * 192;
        int vv = w / 3;
        int c = w - vv * 3;
        lds[SV_OFF + c * SV_PLANE + node * SV_LD + vv] = (bf16)d[j];
      }
    }
  }
  __syncthreads();

  // ---- GEMM1: Vh[c] = V[:, :, c] @ WH^T  (M16, N=128 in 8 tiles, K=64) -----
  v16bf a1[3][2];
#pragma unroll
  for (int c = 0; c < 3; ++c)
#pragma unroll
    for (int kt = 0; kt < 2; ++kt)
      a1[c][kt] = frag_bf(lds + SV_OFF + c * SV_PLANE + m * SV_LD + kt * 32, kh);

#pragma unroll
  for (int ht = 0; ht < 8; ++ht) {
    const bf16* wr = WH + (ht * 16 + col) * 64;
    v16bf b0 = frag_bf(wr, kh);
    v16bf b1 = frag_bf(wr + 32, kh);
    v8f acc0 = {}, acc1 = {}, acc2 = {};
    acc0 = WMMA_BF16(a1[0][0], b0, acc0);  acc0 = WMMA_BF16(a1[0][1], b1, acc0);
    acc1 = WMMA_BF16(a1[1][0], b0, acc1);  acc1 = WMMA_BF16(a1[1][1], b1, acc1);
    acc2 = WMMA_BF16(a1[2][0], b0, acc2);  acc2 = WMMA_BF16(a1[2][1], b1, acc2);
#pragma unroll
    for (int r = 0; r < 8; ++r) {
      int mo = r + 8 * hi;
      int h  = ht * 16 + col;
      float x0 = acc0[r], x1 = acc1[r], x2 = acc2[r];
      lds[VH_OFF + 0 * VH_PLANE + mo * VH_LD + h] = (bf16)x0;
      lds[VH_OFF + 1 * VH_PLANE + mo * VH_LD + h] = (bf16)x1;
      lds[VH_OFF + 2 * VH_PLANE + mo * VH_LD + h] = (bf16)x2;
      lds[NRM_OFF + mo * VH_LD + h] = (bf16)sqrtf(x0 * x0 + x1 * x1 + x2 * x2);  // f32 norm
    }
  }
  __syncthreads();

  // ---- GEMM3: s_out = relu([s | ||Vh||] @ WS^T + bS)  (K=256, N=128) -------
  v16bf a3[8];
#pragma unroll
  for (int kt = 0; kt < 4; ++kt)
    a3[kt] = frag_f32(sIn + (n0 + m) * 128 + kt * 32, kh);
#pragma unroll
  for (int kt = 0; kt < 4; ++kt)
    a3[4 + kt] = frag_bf(lds + NRM_OFF + m * VH_LD + kt * 32, kh);

#pragma unroll
  for (int st = 0; st < 8; ++st) {
    const bf16* wr = WS + (st * 16 + col) * 256;
    v8f acc = {};
#pragma unroll
    for (int kt = 0; kt < 8; ++kt)
      acc = WMMA_BF16(a3[kt], frag_bf(wr + kt * 32, kh), acc);
    float bv = bS[st * 16 + col];
#pragma unroll
    for (int r = 0; r < 8; ++r) {
      float x = acc[r] + bv;
      x = x > 0.f ? x : 0.f;
      sOut[(n0 + r + 8 * hi) * 128 + st * 16 + col] = x;            // output #1
      lds[SO_OFF + (r + 8 * hi) * VH_LD + st * 16 + col] = (bf16)x;
    }
  }
  __syncthreads();

  // ---- GEMM4: gate = sigmoid(s_out @ WG^T + bG)  (K=128, N=64) -------------
  v16bf a4[4];
#pragma unroll
  for (int kt = 0; kt < 4; ++kt)
    a4[kt] = frag_bf(lds + SO_OFF + m * VH_LD + kt * 32, kh);
  v8f gate[4];
#pragma unroll
  for (int ot = 0; ot < 4; ++ot) {
    const bf16* wr = WG + (ot * 16 + col) * 128;
    v8f acc = {};
#pragma unroll
    for (int kt = 0; kt < 4; ++kt)
      acc = WMMA_BF16(a4[kt], frag_bf(wr + kt * 32, kh), acc);
    float bv = bG[ot * 16 + col];
#pragma unroll
    for (int r = 0; r < 8; ++r)
      gate[ot][r] = 1.f / (1.f + expf(-(acc[r] + bv)));
  }

  // ---- GEMM2: V_out[:, :, c] = (Vh[c] @ WV^T) * gate  (K=128, N=64) --------
#pragma unroll
  for (int c = 0; c < 3; ++c) {
    v16bf a2[4];
#pragma unroll
    for (int kt = 0; kt < 4; ++kt)
      a2[kt] = frag_bf(lds + VH_OFF + c * VH_PLANE + m * VH_LD + kt * 32, kh);
#pragma unroll
    for (int ot = 0; ot < 4; ++ot) {
      const bf16* wr = WVm + (ot * 16 + col) * 128;
      v8f acc = {};
#pragma unroll
      for (int kt = 0; kt < 4; ++kt)
        acc = WMMA_BF16(a2[kt], frag_bf(wr + kt * 32, kh), acc);
#pragma unroll
      for (int r = 0; r < 8; ++r)                                  // output #2
        vOut[(n0 + r + 8 * hi) * 192 + (ot * 16 + col) * 3 + c] = acc[r] * gate[ot][r];
    }
  }
}

extern "C" void kernel_launch(void* const* d_in, const int* in_sizes, int n_in,
                              void* d_out, int out_size, void* d_ws, size_t ws_size,
                              hipStream_t stream) {
  const float* s   = (const float*)d_in[0];
  const float* V   = (const float*)d_in[1];
  const float* Wh  = (const float*)d_in[2];
  const float* Wv  = (const float*)d_in[3];
  const float* Wsw = (const float*)d_in[4];
  const float* Wsb = (const float*)d_in[5];
  const float* Wgw = (const float*)d_in[6];
  const float* Wgb = (const float*)d_in[7];
  float* out = (float*)d_out;

  const int N = in_sizes[0] / 128;          // 131072
  bf16* wbf = (bf16*)d_ws;                  // 57344 bf16 = 112 KB scratch

  gvp_wconv<<<224, 256, 0, stream>>>(Wh, Wv, Wsw, Wgw, wbf);

  const int tiles = N / 16;                 // 16-node tiles, one per wave
  const int wgs = tiles / WAVES_PER_WG;
  const size_t ldsBytes = (size_t)WAVES_PER_WG * WAVE_LDS * sizeof(bf16);  // 57344 B
  gvp_fused<<<wgs, 32 * WAVES_PER_WG, ldsBytes, stream>>>(
      s, V, wbf, wbf + 8192, wbf + 16384, wbf + 49152,
      Wsb, Wgb, out, out + (size_t)N * 128);
}
